// LocalAggregator_37701222924997
// MI455X (gfx1250) — matrix operand order repaired and checked by reading, not compiled, not run on hardware
//
#include <hip/hip_runtime.h>
#include <stdint.h>

#define Bdim 32
#define Ndim 256
#define Ddim 128
#define NEG_MASKF -9000000.0f
#define LEAKF 0.2f
#define NBIS 30

typedef __attribute__((ext_vector_type(16))) __bf16 v16bf;
typedef __attribute__((ext_vector_type(8)))  float  v8f;
typedef __attribute__((ext_vector_type(4)))  unsigned int u32x4;
typedef __attribute__((ext_vector_type(8)))  int          i32x8;
typedef __attribute__((ext_vector_type(4)))  int          i32x4;

__device__ __forceinline__ unsigned short f2bf(float f) {
  unsigned u = __float_as_uint(f);
  u += 0x7FFFu + ((u >> 16) & 1u);          // round-to-nearest-even-ish
  return (unsigned short)(u >> 16);
}

__device__ __forceinline__ float wsum(float v) {
#pragma unroll
  for (int o = 16; o > 0; o >>= 1) v += __shfl_xor(v, o, 32);
  return v;
}
__device__ __forceinline__ float wmaxr(float v) {
#pragma unroll
  for (int o = 16; o > 0; o >>= 1) v = fmaxf(v, __shfl_xor(v, o, 32));
  return v;
}
// fast z^inv: reference alphas are 1.5 / 2.0 -> inv = 2 / 1 (wave-uniform branches)
__device__ __forceinline__ float fpow(float z, float inv, bool i1, bool i2) {
  return i1 ? z : (i2 ? z * z : __powf(z, inv));
}

// LDS layout (bytes):
//   h1b  bf16 [256][128]   :      0 ..  65536
//   h2b  bf16 [256][128]   :  65536 .. 131072
//   S1   f32  [ 32][256]   : 131072 .. 163840
//   S2   f32  [ 32][256]   : 163840 .. 196608
//   Wb   bf16 [ 32][256]   : 196608 .. 212992
//   aw   f32  [4][128]     : 212992 .. 215040
//   sw   f32  [4][128]     : 215040 .. 217088
//   adjL i32  [ 32][256]   : 217088 .. 249856   (filled by TDM)
#define ADJ_LDS_OFF 217088
#define LDS_BYTES   249856

__global__ __launch_bounds__(256, 1)
void localagg_fused_kernel(const float* __restrict__ h1g, const float* __restrict__ h2g,
                           const int* __restrict__ adj, const float* __restrict__ awg,
                           const float* __restrict__ swg, const float* __restrict__ a1p,
                           const float* __restrict__ a2p, float* __restrict__ out) {
  extern __shared__ char smem[];
  __bf16* h1b = (__bf16*)(smem);
  __bf16* h2b = (__bf16*)(smem + 65536);
  float*  S1  = (float*)(smem + 131072);
  float*  S2  = (float*)(smem + 163840);
  __bf16* Wb  = (__bf16*)(smem + 196608);
  float*  aw  = (float*)(smem + 212992);
  float*  sw  = (float*)(smem + 215040);
  int*    adjL= (int*)  (smem + ADJ_LDS_OFF);

  const int tid  = threadIdx.x;
  const int wave = tid >> 5;
  const int lane = tid & 31;
  const int b    = blockIdx.x >> 3;
  const int i0   = (blockIdx.x & 7) * 32;     // batch-local row strip [i0, i0+32)

  // ---- TDM: async-copy this block's 32x256 adj strip (contiguous 32KB) into LDS ----
  // D# per CDNA5 ISA: 2D tensor, data_size=4B,
  // tensor_dim0=tile_dim0=8192 elems, tensor_dim1=tile_dim1=1.
  if (wave == 0) {
    const unsigned long long ga =
        (unsigned long long)(uintptr_t)(adj + (size_t)b * Ndim * Ndim + (size_t)i0 * Ndim);
    u32x4 g0 = {1u,                              // count=1, no gather
                (unsigned)ADJ_LDS_OFF,           // lds_addr
                (unsigned)(ga & 0xFFFFFFFFu),    // global_addr[31:0]
                (unsigned)(((ga >> 32) & 0x01FFFFFFu) | (2u << 30))}; // addr[56:32] | type=2
    i32x8 g1 = {0x20000,      // data_size=2 (4 bytes)
                0x20000000,   // tensor_dim0[15:0]=8192 in bits[63:48]
                0x10000,      // tensor_dim1[15:0]=1  in bits[95:80]
                0x20000000,   // tile_dim0=8192       in bits[127:112]
                1,            // tile_dim1=1
                8192,         // tensor_dim0_stride[31:0]
                0x20000000,   // tensor_dim1_stride[15:0]=8192 in bits[223:208]
                0};
    i32x4 gz4 = {0, 0, 0, 0};
    i32x8 gz8 = {0, 0, 0, 0, 0, 0, 0, 0};
    __builtin_amdgcn_tensor_load_to_lds(g0, g1, gz4, gz4, gz8, 0);
  }

  // ---- stage projection weights ----
  for (int t = tid; t < 4 * Ddim; t += 256) { aw[t] = awg[t]; sw[t] = swg[t]; }

  // ---- stage both embeddings of this batch to LDS as bf16 (overlaps the TDM) ----
  {
    const float2* g1v = (const float2*)(h1g + (size_t)b * Ndim * Ddim);
    const float2* g2v = (const float2*)(h2g + (size_t)b * Ndim * Ddim);
    uint32_t* l1 = (uint32_t*)h1b;
    uint32_t* l2 = (uint32_t*)h2b;
    for (int t = tid; t < Ndim * Ddim / 2; t += 256) {
      float2 f1 = g1v[t], f2 = g2v[t];
      l1[t] = (uint32_t)f2bf(f1.x) | ((uint32_t)f2bf(f1.y) << 16);
      l2[t] = (uint32_t)f2bf(f2.x) | ((uint32_t)f2bf(f2.y) << 16);
    }
  }
  if (wave == 0) __builtin_amdgcn_s_wait_tensorcnt(0);
  __syncthreads();

  const int n  = lane & 15;         // WMMA column / A-row lane id
  const int mh = (lane >> 4) & 1;   // high half-lane group

  // ================= Phase 1: masked relation scores (bf16 WMMA) =================
  // wave -> (i-tile, j-chunk): ti in {0,1} (16 rows), j0 in {0,64,128,192}
  const int ti = wave >> 2;
  const int j0 = (wave & 3) * 64;

#pragma unroll 1
  for (int em = 0; em < 2; ++em) {
    const __bf16* hb = em ? h2b : h1b;
    const float*  ww = em ? sw  : aw;
    float* S = em ? S2 : S1;
#pragma unroll 1
    for (int jt = 0; jt < 4; ++jt) {
      const int jb = j0 + jt * 16;
      int adjv[8];
#pragma unroll
      for (int v = 0; v < 8; ++v)
        adjv[v] = adjL[(ti * 16 + mh * 8 + v) * Ndim + jb + n];

      v8f acc[4];
#pragma unroll
      for (int r = 0; r < 4; ++r)
#pragma unroll
        for (int v = 0; v < 8; ++v) acc[r][v] = 0.f;

#pragma unroll
      for (int k0 = 0; k0 < Ddim; k0 += 32) {
        // B fragment: B[K][n] = h[j][K]   (32x16, K in [k0,k0+32))
        v16bf Bf;
        const int kb   = k0 + (mh ? 16 : 0);
        const int jrow = (jb + n) * Ddim;
#pragma unroll
        for (int e = 0; e < 16; ++e) Bf[e] = hb[jrow + kb + e];

        const int ka = k0 + (mh ? 16 : 0);
        const int ia = (i0 + ti * 16 + n) * Ddim;
#pragma unroll
        for (int r = 0; r < 4; ++r) {
          // A fragment: A[M][K] = h[i][K] * w_r[K]  (16x32)
          v16bf Af;
          const float* wr = ww + r * Ddim;
#pragma unroll
          for (int e = 0; e < 16; ++e) {
            const int kk = ka + e;
            Af[e] = (__bf16)((float)hb[ia + kk] * wr[kk]);
          }
          acc[r] = __builtin_amdgcn_wmma_f32_16x16x32_bf16(
              false, Af, false, Bf, (short)0, acc[r], false, false);
        }
      }
      // leaky-relu + select relation per adj, write score strip to LDS
      float sel[8];
#pragma unroll
      for (int v = 0; v < 8; ++v) sel[v] = NEG_MASKF;
#pragma unroll
      for (int r = 0; r < 4; ++r)
#pragma unroll
        for (int v = 0; v < 8; ++v) {
          float e = acc[r][v];
          e = (e >= 0.f) ? e : LEAKF * e;
          if (adjv[v] == r + 1) sel[v] = e;
        }
#pragma unroll
      for (int v = 0; v < 8; ++v)
        S[(ti * 16 + mh * 8 + v) * Ndim + jb + n] = sel[v];
    }
  }
  __syncthreads();

  // ================= Phase 2: entmax-bisect per row (wave32 = one row) =================
  const float alpha1 = a1p[0], alpha2 = a2p[0];
#pragma unroll 1
  for (int t = wave; t < 64; t += 8) {
    float* S = (t & 32) ? S2 : S1;
    const int   row   = t & 31;
    const float alpha = (t & 32) ? alpha2 : alpha1;
    const float am1   = alpha - 1.f;
    const float inv   = 1.f / am1;
    const bool  i1    = (inv == 1.f);   // alpha == 2.0 (sparsemax)
    const bool  i2    = (inv == 2.f);   // alpha == 1.5
    float x[8];
#pragma unroll
    for (int q = 0; q < 8; ++q) x[q] = S[row * Ndim + lane * 8 + q] * am1;
    float mx = x[0];
#pragma unroll
    for (int q = 1; q < 8; ++q) mx = fmaxf(mx, x[q]);
    mx = wmaxr(mx);
    float tau = mx - 1.f;
    const float tau_hi = mx - __powf(1.f / (float)Ndim, am1);
    float s = 0.f;
#pragma unroll
    for (int q = 0; q < 8; ++q) { float z = fmaxf(x[q] - tau, 0.f); s += fpow(z, inv, i1, i2); }
    const float f_lo = wsum(s) - 1.f;
    float dm = tau_hi - tau;
#pragma unroll 1
    for (int it = 0; it < NBIS; ++it) {
      dm *= 0.5f;
      const float tm = tau + dm;
      float fm = 0.f;
#pragma unroll
      for (int q = 0; q < 8; ++q) { float z = fmaxf(x[q] - tm, 0.f); fm += fpow(z, inv, i1, i2); }
      fm = wsum(fm) - 1.f;
      if (fm * f_lo >= 0.f) tau = tm;
    }
    float p[8], sp = 0.f;
#pragma unroll
    for (int q = 0; q < 8; ++q) {
      float z = fmaxf(x[q] - tau, 0.f);
      p[q] = fpow(z, inv, i1, i2); sp += p[q];
    }
    sp = wsum(sp);
    const float rs = 1.f / sp;
#pragma unroll
    for (int q = 0; q < 8; ++q) S[row * Ndim + lane * 8 + q] = p[q] * rs;
  }
  __syncthreads();

  // ---- combine w1*w2, row-normalize, store bf16 weight matrix ----
#pragma unroll 1
  for (int row = wave; row < 32; row += 8) {
    float w[8], s = 0.f;
#pragma unroll
    for (int q = 0; q < 8; ++q) {
      float v = S1[row * Ndim + lane * 8 + q] * S2[row * Ndim + lane * 8 + q];
      w[q] = v; s += v;
    }
    s = wsum(s) + 1e-7f;
    const float rs = 1.f / s;
#pragma unroll
    for (int q = 0; q < 8; ++q) Wb[row * Ndim + lane * 8 + q] = (__bf16)(w[q] * rs);
  }
  __syncthreads();

  // ================= Phase 3: out = weight @ h1 (bf16 WMMA, M=32 K=256 N=128) ========
#pragma unroll 1
  for (int tile = wave; tile < 16; tile += 8) {
    const int mt = tile >> 3;            // 0..1 (16 output rows each)
    const int d0 = (tile & 7) * 16;      // D tile
    v8f acc;
#pragma unroll
    for (int v = 0; v < 8; ++v) acc[v] = 0.f;
#pragma unroll
    for (int k0 = 0; k0 < Ndim; k0 += 32) {
      v16bf Af, Bf;
      const int ka = k0 + (mh ? 16 : 0);
      const int ia = (mt * 16 + n) * Ndim;
#pragma unroll
      for (int e = 0; e < 16; ++e) {
        const int kk = ka + e;
        Af[e] = Wb[ia + kk];
      }
      const int kb = k0 + (mh ? 16 : 0);
#pragma unroll
      for (int e = 0; e < 16; ++e) Bf[e] = h1b[(kb + e) * Ddim + d0 + n];
      acc = __builtin_amdgcn_wmma_f32_16x16x32_bf16(
          false, Af, false, Bf, (short)0, acc, false, false);
    }
#pragma unroll
    for (int v = 0; v < 8; ++v)
      out[((size_t)b * Ndim + i0 + mt * 16 + mh * 8 + v) * Ddim + d0 + n] = acc[v];
  }
}

extern "C" void kernel_launch(void* const* d_in, const int* in_sizes, int n_in,
                              void* d_out, int out_size, void* d_ws, size_t ws_size,
                              hipStream_t stream) {
  const float* h1 = (const float*)d_in[0];
  const float* h2 = (const float*)d_in[1];
  const int*  adj = (const int*)d_in[2];
  const float* aw = (const float*)d_in[3];
  const float* sw = (const float*)d_in[4];
  const float* a1 = (const float*)d_in[5];
  const float* a2 = (const float*)d_in[6];
  float* out = (float*)d_out;

  (void)hipFuncSetAttribute(reinterpret_cast<const void*>(localagg_fused_kernel),
                            hipFuncAttributeMaxDynamicSharedMemorySize, LDS_BYTES);
  // 32 batches * 8 row-strips, 256 threads (8 wave32) each
  localagg_fused_kernel<<<dim3(Bdim * 8), dim3(256), LDS_BYTES, stream>>>(
      h1, h2, adj, aw, sw, a1, a2, out);
}
